// MutationCrossEntropyLoss_26903675142293
// MI455X (gfx1250) — matrix operand order repaired and checked by reading, not compiled, not run on hardware
//
#include <hip/hip_runtime.h>

typedef __attribute__((ext_vector_type(16))) _Float16 v16h;
typedef __attribute__((ext_vector_type(8)))  float    v8f;

#define C_DIM           30
#define ROWS_PER_BLOCK  256
#define WAVES_PER_BLOCK 8
#define XS_STRIDE       33   // pad 32->33 dwords: conflict-free strided LDS reads

__global__ void mce_zero_out(float* out) {
    if (threadIdx.x == 0) out[0] = 0.0f;
}

__global__ __launch_bounds__(256) void mce_loss_kernel(
    const float* __restrict__ inp,      // (N, 30) fp32
    const long long* __restrict__ tgt,  // (N,) int64
    const float* __restrict__ Mmat,     // (30, 30) fp32
    float* __restrict__ out,            // scalar
    int N, float invN)
{
    __shared__ float Ms[30 * 33];                        // padded mutation matrix
    __shared__ float rowsumM[32];
    __shared__ float Xs[WAVES_PER_BLOCK][32 * XS_STRIDE]; // X tile; reused as D spill
    __shared__ float redbuf[WAVES_PER_BLOCK];

    const int tid  = threadIdx.x;
    const int wave = tid >> 5;
    const int lane = tid & 31;

    // ---- load M into LDS (padded), compute per-class row sums ----
    for (int i = tid; i < 30 * 30; i += 256)
        Ms[(i / 30) * 33 + (i % 30)] = Mmat[i];
    __syncthreads();
    if (tid < 30) {
        float s = 0.0f;
        #pragma unroll
        for (int j = 0; j < 30; ++j) s += Ms[tid * 33 + j];
        rowsumM[tid] = s;
    }

    // ---- stream this lane's row: 15 x b64 loads (row stride 120B, 8B aligned) ----
    const int row = blockIdx.x * ROWS_PER_BLOCK + tid;
    const int rr  = (row < N) ? row : (N - 1);
    const float* xrow = inp + (long long)rr * C_DIM;
    float xr[30];
    #pragma unroll
    for (int j = 0; j < 15; ++j) {
        float2 v = ((const float2*)xrow)[j];
        xr[2 * j]     = v.x;
        xr[2 * j + 1] = v.y;
    }

    // ---- fp32 numerically-stable logsumexp in registers ----
    float mx = xr[0];
    #pragma unroll
    for (int j = 1; j < 30; ++j) mx = fmaxf(mx, xr[j]);
    float es = 0.0f;
    #pragma unroll
    for (int j = 0; j < 30; ++j) es += __expf(xr[j] - mx);
    const float lse = mx + __logf(es);

    // ---- target ----
    const long long t64 = tgt[rr];
    int t = (int)t64;
    t = (t < 0) ? 0 : ((t > 29) ? 29 : t);
    const bool valid = (row < N) && (t64 != 0);

    // ---- stage row into LDS X tile (K padded to 32 with zeros) ----
    float* xw = &Xs[wave][lane * XS_STRIDE];
    #pragma unroll
    for (int j = 0; j < 30; ++j) xw[j] = xr[j];
    xw[30] = 0.0f;
    xw[31] = 0.0f;
    __syncthreads();

    // ---- build B = M^T tiles (32x16 f16), loop-invariant, classes 0..15 / 16..29 ----
    // B layout: lanes 0-15 hold K={0..7,16..23}, lanes 16-31 hold K={8..15,24..31}; column N = lane%16
    v16h b0, b1;
    {
        const int n     = lane & 15;
        const int kbase = (lane < 16) ? 0 : 8;
        #pragma unroll
        for (int e = 0; e < 16; ++e) {
            const int k  = kbase + ((e < 8) ? e : e + 8);
            const int c1 = 16 + n;
            const float v0 = (k < 30) ? Ms[n * 33 + k] : 0.0f;
            const float v1 = (k < 30 && c1 < 30) ? Ms[c1 * 33 + k] : 0.0f;
            b0[e] = (_Float16)v0;
            b1[e] = (_Float16)v1;
        }
    }

    // ---- build A tiles (16x32 f16) per ISA layout:
    // lanes 0-15: K = {0..7, 16..23}; lanes 16-31: K = {8..15, 24..31}
    v16h a0, a1;
    {
        const int m    = lane & 15;
        const int koff = (lane < 16) ? 0 : 8;
        const float* x0 = &Xs[wave][m * XS_STRIDE];
        const float* x1 = &Xs[wave][(16 + m) * XS_STRIDE];
        #pragma unroll
        for (int e = 0; e < 16; ++e) {
            const int k = ((e < 8) ? e : e + 8) + koff;
            a0[e] = (_Float16)x0[k];
            a1[e] = (_Float16)x1[k];
        }
    }
    __syncthreads();  // all Xs reads done before reuse as D spill

    // ---- WMMA: D[m,n] = dot(x_m, M[n + nb, :]) ----
    v8f d00 = {}, d01 = {}, d10 = {}, d11 = {};
    d00 = __builtin_amdgcn_wmma_f32_16x16x32_f16(false, a0, false, b0, (short)0, d00, false, false);
    d01 = __builtin_amdgcn_wmma_f32_16x16x32_f16(false, a0, false, b1, (short)0, d01, false, false);
    d10 = __builtin_amdgcn_wmma_f32_16x16x32_f16(false, a1, false, b0, (short)0, d10, false, false);
    d11 = __builtin_amdgcn_wmma_f32_16x16x32_f16(false, a1, false, b1, (short)0, d11, false, false);

    // ---- spill D into LDS (reuse Xs[wave]): D layout VGPR r -> M=r (lanes<16) / r+8 ----
    float* Dr = &Xs[wave][0];
    {
        const int n    = lane & 15;
        const int mofs = (lane < 16) ? 0 : 8;
        #pragma unroll
        for (int r = 0; r < 8; ++r) {
            const int m = r + mofs;
            Dr[0   + m * 16 + n] = d00[r];
            Dr[256 + m * 16 + n] = d01[r];
            Dr[512 + m * 16 + n] = d10[r];
            Dr[768 + m * 16 + n] = d11[r];
        }
    }
    __syncthreads();

    // ---- each lane gathers its row's dot(M[t,:], x): tile = lane/16, class block = t/16 ----
    const int T   = lane >> 4;
    const int m   = lane & 15;
    const int mat = T * 2 + ((t >= 16) ? 1 : 0);
    const float dot = Dr[mat * 256 + m * 16 + (t & 15)];

    float contrib = valid ? (rowsumM[t] * lse - dot) : 0.0f;

    // ---- reduce: shfl tree within wave, LDS across waves, 1 atomic per block ----
    #pragma unroll
    for (int off = 16; off > 0; off >>= 1)
        contrib += __shfl_down(contrib, off, 32);
    if (lane == 0) redbuf[wave] = contrib;
    __syncthreads();
    if (tid == 0) {
        float tot = 0.0f;
        #pragma unroll
        for (int w = 0; w < WAVES_PER_BLOCK; ++w) tot += redbuf[w];
        atomicAdd(out, tot * invN);
    }
}

extern "C" void kernel_launch(void* const* d_in, const int* in_sizes, int n_in,
                              void* d_out, int out_size, void* d_ws, size_t ws_size,
                              hipStream_t stream) {
    const float*     inp  = (const float*)d_in[0];
    const long long* tgt  = (const long long*)d_in[1];
    const float*     Mmat = (const float*)d_in[2];
    float*           out  = (float*)d_out;

    const int N = in_sizes[1];                 // 4194304
    const int blocks = (N + ROWS_PER_BLOCK - 1) / ROWS_PER_BLOCK;

    hipLaunchKernelGGL(mce_zero_out, dim3(1), dim3(64), 0, stream, out);
    hipLaunchKernelGGL(mce_loss_kernel, dim3(blocks), dim3(256), 0, stream,
                       inp, tgt, Mmat, out, N, 1.0f / (float)N);
}
